// AllegroEmbedding_56109452755158
// MI455X (gfx1250) — compile-verified, matrix operand order checked by reading, not run on hardware
//
#include <hip/hip_runtime.h>
#include <hip/hip_bf16.h>

typedef __attribute__((ext_vector_type(16))) _Float16 v16h;
typedef __attribute__((ext_vector_type(8)))  _Float16 h8;
typedef __attribute__((ext_vector_type(8)))  float    v8f;

union V16U { v16h v; h8 h[2]; };

#define TILE_M   128
#define NTHREADS 256

// ---- LDS plan (bytes):
// weight frags (f16, B-fragment swizzled):
//   wf0 96x64=12K, wf1 64x128=16K, wf2 128x256=64K, wf3 256x64=32K, wf4 64x80=10K  -> 134K
// activation ping-pong: buf0 128x128 f16 = 32K, buf1 128x256 f16 = 64K            -> 96K
// env 0.5K, Y 4.5K, es||er f32 32K                                                -> 37K
// total ~267 KB  (< 320 KB per WGP)
struct SMem {
  alignas(16) _Float16 wf0[96 * 64];
  alignas(16) _Float16 wf1[64 * 128];
  alignas(16) _Float16 wf2[128 * 256];
  alignas(16) _Float16 wf3[256 * 64];
  alignas(16) _Float16 wf4[64 * 80];
  alignas(16) _Float16 buf0[TILE_M * 128];
  alignas(16) _Float16 buf1[TILE_M * 256];
  float env[TILE_M];
  float Yl[TILE_M * 9];
  float eser[TILE_M * 64];
};

// Convert a row-major f32 weight (Kact x Nact) into f16 B-fragment layout in LDS,
// padded to K x N.  Fragment layout per 16x16x32 WMMA B operand (ISA 7.12.2):
// lane l covers column n = nt*16 + (l&15); its 16 halfs cover
// K = kt*32 + base + i + (i>=8 ? 8 : 0), base = (l>>4)*8.
__device__ __forceinline__ void fill_wfrag(const float* __restrict__ W,
                                           int K, int N, int Kact, int Nact,
                                           _Float16* __restrict__ dst, int tid) {
  const int NT = N >> 4;
  const int total = K * N;
  for (int idx = tid; idx < total; idx += NTHREADS) {
    const int i    = idx & 15;
    const int lane = (idx >> 4) & 31;
    const int blk  = idx >> 9;          // kt*NT + nt
    const int nt   = blk % NT;
    const int kt   = blk / NT;
    const int n    = nt * 16 + (lane & 15);
    const int base = (lane >> 4) << 3;
    const int k    = kt * 32 + base + i + ((i >> 3) << 3);
    float v = 0.0f;
    if (k < Kact && n < Nact) v = W[k * Nact + n];
    dst[idx] = (_Float16)v;
  }
}

// One wave computes its 16-row slice times the weight block: acc[nt] += A(16x32) x B(32x16).
// A read from row-major f16 LDS activations; B read from pre-swizzled fragments.
template <int KT, int NTC>
__device__ __forceinline__ void wave_gemm(const _Float16* __restrict__ actIn, int K,
                                          const _Float16* __restrict__ wf, int NT, int ntBegin,
                                          int lane, int row, int base, v8f acc[NTC]) {
#pragma unroll
  for (int kt = 0; kt < KT; ++kt) {
    V16U a;
    const int c0 = kt * 32 + base;
    a.h[0] = *(const h8*)(actIn + row * K + c0);
    a.h[1] = *(const h8*)(actIn + row * K + c0 + 16);
#pragma unroll
    for (int nt = 0; nt < NTC; ++nt) {
      V16U b;
      const _Float16* bp = wf + ((size_t)((kt * NT) + ntBegin + nt) * 32 + lane) * 16;
      b.h[0] = *(const h8*)bp;
      b.h[1] = *(const h8*)(bp + 8);
      acc[nt] = __builtin_amdgcn_wmma_f32_16x16x32_f16(
          false, a.v, false, b.v, (short)0, acc[nt], false, false);
    }
  }
}

template <int NTC>
__device__ __forceinline__ void store_silu(const v8f acc[NTC], int ntBegin, float scale,
                                           _Float16* __restrict__ dst, int W,
                                           int hi, int ncol, int waveM0) {
#pragma unroll
  for (int nt = 0; nt < NTC; ++nt) {
#pragma unroll
    for (int v = 0; v < 8; ++v) {
      const int rr = waveM0 + hi * 8 + v;   // C/D layout: VGPR v -> row v (+8 for hi half)
      const float x = acc[nt][v] * scale;
      const float s = x / (1.0f + __expf(-x));
      dst[rr * W + (ntBegin + nt) * 16 + ncol] = (_Float16)s;
    }
  }
}

__global__ __launch_bounds__(NTHREADS, 1)
void allegro_fused_kernel(const float* __restrict__ vectors,
                          const int*   __restrict__ senders,
                          const int*   __restrict__ receivers,
                          const int*   __restrict__ species,
                          const float* __restrict__ embed,
                          const float* __restrict__ freqs,
                          const float* __restrict__ W0, const float* __restrict__ W1,
                          const float* __restrict__ W2, const float* __restrict__ W3,
                          const float* __restrict__ Ww,
                          float* __restrict__ out, int E) {
  __shared__ SMem sm;
  const int tid    = threadIdx.x;
  const int lane   = tid & 31;
  const int wave   = tid >> 5;
  const int waveM0 = wave * 16;
  const int e0     = blockIdx.x * TILE_M;

  // ---- phase 0: stage weights into LDS as WMMA B-fragments (f32 -> f16) ----
  fill_wfrag(W0,  96,  64,  72,  64, sm.wf0, tid);
  fill_wfrag(W1,  64, 128,  64, 128, sm.wf1, tid);
  fill_wfrag(W2, 128, 256, 128, 256, sm.wf2, tid);
  fill_wfrag(W3, 256,  64, 256,  64, sm.wf3, tid);
  fill_wfrag(Ww,  64,  80,  64,  67, sm.wf4, tid);

  // ---- phase 1: per-edge featurization -> X0 (f16, width 96, zero padded) ----
  if (tid < TILE_M) {
    const int  r     = tid;
    const int  e     = e0 + r;
    const bool valid = e < E;
    const int  ec    = valid ? e : (E - 1);
    const float vx = vectors[3 * ec + 0];
    const float vy = vectors[3 * ec + 1];
    const float vz = vectors[3 * ec + 2];
    const float d  = sqrtf(vx * vx + vy * vy + vz * vz);
    const float d2 = d * d;
    const float d6 = d2 * d2 * d2, d7 = d6 * d, d8 = d7 * d;
    float env = (d < 1.0f) ? (1.0f - 28.0f * d6 + 48.0f * d7 - 21.0f * d8) : 0.0f;
    if (!valid) env = 0.0f;
    sm.env[r] = env;
    const float invd = 1.0f / d;
    const float ux = vx * invd, uy = vy * invd, uz = vz * invd;
    const float s3c = 1.7320508075688772f;   // sqrt(3)
    const float s15 = 3.8729833462074170f;   // sqrt(15)
    const float s5h = 1.1180339887498949f;   // 0.5*sqrt(5)
    float Y[9];
    Y[0] = 1.0f;
    Y[1] = s3c * ux; Y[2] = s3c * uy; Y[3] = s3c * uz;
    Y[4] = s15 * ux * uy; Y[5] = s15 * uy * uz;
    Y[6] = s5h * (3.0f * uz * uz - 1.0f);
    Y[7] = s15 * ux * uz; Y[8] = 0.5f * s15 * (ux * ux - uy * uy);
#pragma unroll
    for (int k = 0; k < 9; ++k) sm.Yl[r * 9 + k] = Y[k];
    const float sq2 = 1.4142135623730951f;
#pragma unroll
    for (int k = 0; k < 8; ++k) {
      const float rb = valid ? (sq2 * __sinf(freqs[k] * d) * invd * env) : 0.0f;
      sm.buf0[r * 96 + k] = (_Float16)rb;
    }
#pragma unroll
    for (int k = 72; k < 96; ++k) sm.buf0[r * 96 + k] = (_Float16)0.0f;
  }
  // species-embedding gathers (sender||receiver), also kept f32 for V epilogue
  for (int idx = tid; idx < TILE_M * 64; idx += NTHREADS) {
    const int  r     = idx >> 6;
    const int  j     = idx & 63;
    const int  e     = e0 + r;
    const bool valid = e < E;
    const int  ec    = valid ? e : (E - 1);
    const int  node  = (j < 32) ? senders[ec] : receivers[ec];
    float v = embed[species[node] * 32 + (j & 31)];
    if (!valid) v = 0.0f;
    sm.eser[r * 64 + j]    = v;
    sm.buf0[r * 96 + 8 + j] = (_Float16)v;
  }
  __syncthreads();

  const int base = (lane >> 4) * 8;     // A-frag K sub-block per lane half
  const int row  = waveM0 + (lane & 15);
  const int hi   = lane >> 4;
  const int ncol = lane & 15;

  // ---- L0: 96 -> 64, silu  (scale 1/sqrt(72)) ----
  {
    v8f acc[4] = {};
    wave_gemm<3, 4>(sm.buf0, 96, sm.wf0, 4, 0, lane, row, base, acc);
    store_silu<4>(acc, 0, 0.11785113019775793f, sm.buf1, 64, hi, ncol, waveM0);
  }
  __syncthreads();
  // ---- L1: 64 -> 128, silu  (scale 1/8) ----
  {
    v8f acc[8] = {};
    wave_gemm<2, 8>(sm.buf1, 64, sm.wf1, 8, 0, lane, row, base, acc);
    store_silu<8>(acc, 0, 0.125f, sm.buf0, 128, hi, ncol, waveM0);
  }
  __syncthreads();
  // ---- L2: 128 -> 256, silu (scale 1/sqrt(128)); two column passes ----
  {
    v8f acc[8] = {};
    wave_gemm<4, 8>(sm.buf0, 128, sm.wf2, 16, 0, lane, row, base, acc);
    store_silu<8>(acc, 0, 0.08838834764831845f, sm.buf1, 256, hi, ncol, waveM0);
    v8f acc2[8] = {};
    wave_gemm<4, 8>(sm.buf0, 128, sm.wf2, 16, 8, lane, row, base, acc2);
    store_silu<8>(acc2, 8, 0.08838834764831845f, sm.buf1, 256, hi, ncol, waveM0);
  }
  __syncthreads();
  // ---- L3: 256 -> 64 (scale 1/16), envelope scale, emit x ----
  {
    v8f acc[4] = {};
    wave_gemm<8, 4>(sm.buf1, 256, sm.wf3, 4, 0, lane, row, base, acc);
#pragma unroll
    for (int nt = 0; nt < 4; ++nt) {
#pragma unroll
      for (int v = 0; v < 8; ++v) {
        const int rr  = waveM0 + hi * 8 + v;
        const int col = nt * 16 + ncol;
        const float xv = sm.env[rr] * (acc[nt][v] * 0.0625f);
        if (e0 + rr < E) out[(size_t)(e0 + rr) * 64 + col] = xv;
        sm.buf0[rr * 64 + col] = (_Float16)xv;
      }
    }
  }
  __syncthreads();
  // ---- L4: 64 -> 67 (padded 80, scale 1/8); irrep-repeat, emit V ----
  {
    v8f acc[5] = {};
    wave_gemm<2, 5>(sm.buf0, 64, sm.wf4, 5, 0, lane, row, base, acc);
    const float invNI = 1.0f / 67.0f;
    float* __restrict__ Vout = out + (size_t)E * 64;
#pragma unroll
    for (int nt = 0; nt < 5; ++nt) {
      const int n = nt * 16 + ncol;
      if (n < 67) {
#pragma unroll
        for (int v = 0; v < 8; ++v) {
          const int rr = waveM0 + hi * 8 + v;
          const int e  = e0 + rr;
          if (e < E) {
            const float w = acc[nt][v] * 0.125f * invNI;
            float* __restrict__ vp = Vout + (size_t)e * 73;
            if (n == 0) {
              vp[0] = w * sm.Yl[rr * 9 + 0];
            } else if (n == 1) {
              vp[1] = w * sm.Yl[rr * 9 + 1];
              vp[2] = w * sm.Yl[rr * 9 + 2];
              vp[3] = w * sm.Yl[rr * 9 + 3];
            } else if (n == 2) {
#pragma unroll
              for (int q = 0; q < 5; ++q) vp[4 + q] = w * sm.Yl[rr * 9 + 4 + q];
            } else {
              vp[9 + (n - 3)] = w * sm.eser[rr * 64 + (n - 3)];
            }
          }
        }
      }
    }
  }
}

extern "C" void kernel_launch(void* const* d_in, const int* in_sizes, int n_in,
                              void* d_out, int out_size, void* d_ws, size_t ws_size,
                              hipStream_t stream) {
  const float* vectors   = (const float*)d_in[0];
  const int*   senders   = (const int*)d_in[1];
  const int*   receivers = (const int*)d_in[2];
  const int*   species   = (const int*)d_in[3];
  const float* embed     = (const float*)d_in[4];
  const float* freqs     = (const float*)d_in[5];
  const float* W0        = (const float*)d_in[6];
  const float* W1        = (const float*)d_in[7];
  const float* W2        = (const float*)d_in[8];
  const float* W3        = (const float*)d_in[9];
  const float* Ww        = (const float*)d_in[10];
  float*       out       = (float*)d_out;
  const int E = in_sizes[1];          // senders count
  const int grid = (E + TILE_M - 1) / TILE_M;
  allegro_fused_kernel<<<dim3(grid), dim3(NTHREADS), 0, stream>>>(
      vectors, senders, receivers, species, embed, freqs, W0, W1, W2, W3, Ww, out, E);
}
